// Lfm2ShortConv_68470368633593
// MI455X (gfx1250) — matrix operand order repaired and checked by reading, not compiled, not run on hardware
//
#include <hip/hip_runtime.h>

typedef __attribute__((ext_vector_type(16))) __bf16        v16bf;
typedef __attribute__((ext_vector_type(8)))  float         v8f;
typedef __attribute__((ext_vector_type(4)))  unsigned int  u32x4;
typedef __attribute__((ext_vector_type(8)))  int           i32x8;
typedef __attribute__((ext_vector_type(4)))  int           i32x4;

#define BB 2048   // batch
#define DD 1024   // hidden dim (K and N of both GEMMs)
#define LC 3      // conv cache taps

__device__ __forceinline__ unsigned short f2bf(float f) {
  unsigned int u = __float_as_uint(f);
  u += 0x7FFFu + ((u >> 16) & 1u);           // round-to-nearest-even
  return (unsigned short)(u >> 16);
}

union FragBF { v16bf v; u32x4 q[2]; };

// Issue a TDM 2D tile load: rows=128, cols=32 bf16 elements, row stride = K
// elements in memory, +16B LDS padding per 64B row (-> 80B LDS row stride).
__device__ __forceinline__ void tdm_load_tile(const unsigned short* gbase,
                                              unsigned lds_byte_off,
                                              unsigned rows_total) {
  const unsigned long long ga = (unsigned long long)(uintptr_t)gbase;
  u32x4 g0;
  g0.x = 1u;                                               // count=1 (valid)
  g0.y = lds_byte_off;                                     // D#.lds_addr
  g0.z = (unsigned)ga;                                     // global_addr[31:0]
  g0.w = ((unsigned)(ga >> 32) & 0x01FFFFFFu) | 0x80000000u; // [56:32] | type=2
  i32x8 g1;
  // data_size=2B (code1), pad_enable=1, pad_interval=3 (16 DW = 64B rows),
  // pad_amount=3 (4 DW = 16B)
  g1[0] = 0x06D10000;
  g1[1] = (int)((DD & 0xFFFFu) << 16);                     // tensor_dim0 = K
  g1[2] = (int)((rows_total & 0xFFFFu) << 16);             // tensor_dim1
  g1[3] = (32 << 16);                                      // tile_dim0 = 32
  g1[4] = 128;                                             // tile_dim1 = 128
  g1[5] = DD;                                              // tensor_dim0_stride
  g1[6] = 0;
  g1[7] = 0;
  i32x4 z4 = {0, 0, 0, 0};
#if __clang_major__ >= 23
  i32x8 z8 = {0, 0, 0, 0, 0, 0, 0, 0};
  __builtin_amdgcn_tensor_load_to_lds(g0, g1, z4, z4, z8, 0);
#else
  __builtin_amdgcn_tensor_load_to_lds(g0, g1, z4, z4, 0);
#endif
}

// C[m0..+128][n0..+128] = A[M][K] * Bw[N][K]^T   (A, Bw bf16 row-major, K = DD)
// Block = 256 threads = 8 waves (2Mx4N); wave tile 64x32; TDM-fed double-
// buffered LDS (wave 0 drives the DMA, TENSORcnt-synchronized).
__device__ __forceinline__ void gemm_mainloop(
    const unsigned short* __restrict__ A,
    const unsigned short* __restrict__ Bw,
    int m0, int n0, v8f acc[4][2]) {
  constexpr int K    = DD;
  constexpr int LDSS = 40;        // 32 + 8-element pad (matches TDM padding)
  constexpr int BUF_EL = 10240;   // elements per double-buffer slot (A+B)
  constexpr int BUF_BY = 20480;   // bytes per slot
  extern __shared__ unsigned short smem[];  // 2 * 20480 B = 40 KB dynamic LDS

  const int tid  = threadIdx.x;
  const int lane = tid & 31;
  const int wave = tid >> 5;
  const int wm   = (wave >> 2) * 64;
  const int wn   = (wave & 3) * 32;
  const int half = lane >> 4;     // lanes 16-31 hold K 8-15 and 24-31
  const int l15  = lane & 15;

  const unsigned short* Abase = A  + (size_t)m0 * K;
  const unsigned short* Bbase = Bw + (size_t)n0 * K;

  // Prologue: DMA tiles for k0 = 0 into buffer 0.
  if (wave == 0) {
    tdm_load_tile(Abase, 0, BB);
    tdm_load_tile(Bbase, 10240, DD);
  }

  int cur = 0;
  for (int k0 = 0; k0 < K; k0 += 32) {
    if (wave == 0) {
      if (k0 + 32 < K) {
        // Issue next tiles into the other buffer, then wait for current two.
        tdm_load_tile(Abase + (k0 + 32), (unsigned)((cur ^ 1) * BUF_BY), BB);
        tdm_load_tile(Bbase + (k0 + 32), (unsigned)((cur ^ 1) * BUF_BY + 10240), DD);
        __builtin_amdgcn_s_wait_tensorcnt(2);
      } else {
        __builtin_amdgcn_s_wait_tensorcnt(0);
      }
    }
    __syncthreads();   // release waves 1-7 once tiles have landed

    const unsigned short* sA = smem + cur * BUF_EL;
    const unsigned short* sB = smem + cur * BUF_EL + 5120;

    #pragma unroll
    for (int u = 0; u < 2; ++u) {
      // B fragment (32x16): row n holds K contiguously; lane takes K octets
      // at half*8 and 16+half*8.
      FragBF bf;
      const int n = wn + u * 16 + l15;
      bf.q[0] = *(const u32x4*)(&sB[n * LDSS + half * 8]);
      bf.q[1] = *(const u32x4*)(&sB[n * LDSS + 16 + half * 8]);
      #pragma unroll
      for (int t = 0; t < 4; ++t) {
        // A fragment (16x32): lane holds row m, K octets at half*8, 16+half*8.
        FragBF af;
        const int m = wm + t * 16 + l15;
        af.q[0] = *(const u32x4*)(&sA[m * LDSS + half * 8]);
        af.q[1] = *(const u32x4*)(&sA[m * LDSS + 16 + half * 8]);
        acc[t][u] = __builtin_amdgcn_wmma_f32_16x16x32_bf16(
            false, af.v, false, bf.v, (short)0, acc[t][u], false, false);
      }
    }
    __syncthreads();   // all reads done before next DMA overwrites this buffer
    cur ^= 1;
  }
}

// ---- Kernel 1: convert x, in_proj_w[D:2D,:], out_proj_w to bf16 ----
__global__ void prep_bf16(const float* __restrict__ x,
                          const float* __restrict__ in_proj,
                          const float* __restrict__ out_proj,
                          unsigned short* __restrict__ Xb,
                          unsigned short* __restrict__ Wc,
                          unsigned short* __restrict__ Wo) {
  const int total = BB * DD + 2 * DD * DD;
  for (int i = blockIdx.x * blockDim.x + threadIdx.x; i < total;
       i += gridDim.x * blockDim.x) {
    if (i < BB * DD) {
      Xb[i] = f2bf(x[i]);
    } else if (i < BB * DD + DD * DD) {
      const int j = i - BB * DD;
      Wc[j] = f2bf(in_proj[(size_t)DD * DD + j]);  // rows D..2D-1 (the C gate)
    } else {
      const int j = i - BB * DD - DD * DD;
      Wo[j] = f2bf(out_proj[j]);
    }
  }
}

// ---- Kernel 2: Cg = Xb @ Wc^T, fused with 3-tap conv gate; bf16 T out ----
__global__ void __launch_bounds__(256)
gemm1_conv_gate(const unsigned short* __restrict__ Xb,
                const unsigned short* __restrict__ Wc,
                const float* __restrict__ conv_cache,   // layer 0 at offset 0
                const float* __restrict__ conv_w,       // [D,1,LC]
                unsigned short* __restrict__ T) {
  const int n0 = blockIdx.x * 128;
  const int m0 = blockIdx.y * 128;
  v8f acc[4][2] = {};
  gemm_mainloop(Xb, Wc, m0, n0, acc);

  const int lane = threadIdx.x & 31;
  const int wave = threadIdx.x >> 5;
  const int wm = (wave >> 2) * 64, wn = (wave & 3) * 32;
  const int half = lane >> 4, l15 = lane & 15;

  #pragma unroll
  for (int u = 0; u < 2; ++u) {
    const int gn = n0 + wn + u * 16 + l15;     // channel d (fixed per lane)
    const float w0 = conv_w[gn * LC + 0];
    const float w1 = conv_w[gn * LC + 1];
    const float w2 = conv_w[gn * LC + 2];
    #pragma unroll
    for (int t = 0; t < 4; ++t) {
      #pragma unroll
      for (int r = 0; r < 8; ++r) {
        const int gm = m0 + wm + t * 16 + half * 8 + r;   // batch b
        const size_t cb = ((size_t)gm * DD + gn) * LC;
        const float g = conv_cache[cb] * w0 + conv_cache[cb + 1] * w1 +
                        conv_cache[cb + 2] * w2;
        T[(size_t)gm * DD + gn] = f2bf(acc[t][u][r] * g);
      }
    }
  }
}

// ---- Kernel 3: out = T @ Wo^T (f32 output) ----
__global__ void __launch_bounds__(256)
gemm2_out(const unsigned short* __restrict__ T,
          const unsigned short* __restrict__ Wo,
          float* __restrict__ out) {
  const int n0 = blockIdx.x * 128;
  const int m0 = blockIdx.y * 128;
  v8f acc[4][2] = {};
  gemm_mainloop(T, Wo, m0, n0, acc);

  const int lane = threadIdx.x & 31;
  const int wave = threadIdx.x >> 5;
  const int wm = (wave >> 2) * 64, wn = (wave & 3) * 32;
  const int half = lane >> 4, l15 = lane & 15;

  #pragma unroll
  for (int u = 0; u < 2; ++u) {
    const int gn = n0 + wn + u * 16 + l15;
    #pragma unroll
    for (int t = 0; t < 4; ++t) {
      #pragma unroll
      for (int r = 0; r < 8; ++r) {
        const int gm = m0 + wm + t * 16 + half * 8 + r;
        out[(size_t)gm * DD + gn] = acc[t][u][r];
      }
    }
  }
}

extern "C" void kernel_launch(void* const* d_in, const int* in_sizes, int n_in,
                              void* d_out, int out_size, void* d_ws, size_t ws_size,
                              hipStream_t stream) {
  const float* x  = (const float*)d_in[0];   // [B,1,D]
  const float* cc = (const float*)d_in[1];   // [12,B,D,LC]; layer 0 at offset 0
  // d_in[2]: cache_position — always > 0 in this workload; WithCache branch taken
  const float* wi = (const float*)d_in[3];   // [3D,D]
  const float* wo = (const float*)d_in[4];   // [D,D]
  const float* cw = (const float*)d_in[5];   // [D,1,LC]

  unsigned short* Xb = (unsigned short*)d_ws;             // 2048*1024 bf16
  unsigned short* Wc = Xb + (size_t)BB * DD;              // 1024*1024 bf16
  unsigned short* Wo = Wc + (size_t)DD * DD;              // 1024*1024 bf16
  unsigned short* T  = Wo + (size_t)DD * DD;              // 2048*1024 bf16
  float* out = (float*)d_out;

  prep_bf16<<<2048, 256, 0, stream>>>(x, wi, wo, Xb, Wc, Wo);
  dim3 grid(DD / 128, BB / 128);
  const size_t smem_bytes = 2 * 20480;   // double-buffered A+B tiles
  gemm1_conv_gate<<<grid, 256, smem_bytes, stream>>>(Xb, Wc, cc, cw, T);
  gemm2_out<<<grid, 256, smem_bytes, stream>>>(T, Wo, out);
}
